// GraphAttentionEncoder_10187662426147
// MI455X (gfx1250) — matrix operand order repaired and checked by reading, not compiled, not run on hardware
//
#include <hip/hip_runtime.h>
#include <hip/hip_bf16.h>
#include <hip/hip_fp16.h>

typedef __attribute__((ext_vector_type(16))) _Float16 v16h;
typedef __attribute__((ext_vector_type(8)))  float    v8f;

#define BB   32
#define NNQ  512
#define DD   512
#define HHN  8
#define DKK  64
#define LLN  3
#define FFD  2048
#define TOK  (BB*NNQ)   /* 16384 tokens */

union FragU { uint4 u4[2]; v16h h; };

// A-matrix 16x32 f16 fragment (rows across lanes, interleaved K halves):
// lane<16: row=lane, halves K{0..7,16..23}; lane>=16: row=lane-16, K{8..15,24..31}
__device__ __forceinline__ v16h frag_ld_a(const _Float16* p0, int strideH) {
  int lane = threadIdx.x & 31;
  const _Float16* p = p0 + (size_t)(lane & 15) * strideH + ((lane >> 4) << 3);
  FragU f;
  f.u4[0] = *(const uint4*)(p);
  f.u4[1] = *(const uint4*)(p + 16);
  return f.h;
}
// B-matrix 32x16 f16 fragment from Bt stored row-major [N][K]:
// lane<16: col=lane, K{0..7,16..23}; lane>=16: col=lane-16, K{8..15,24..31}
__device__ __forceinline__ v16h frag_ld_b(const _Float16* p0, int strideH) {
  int lane = threadIdx.x & 31;
  const _Float16* p = p0 + (size_t)(lane & 15) * strideH + ((lane >> 4) << 3);
  FragU f;
  f.u4[0] = *(const uint4*)(p);
  f.u4[1] = *(const uint4*)(p + 16);
  return f.h;
}
__device__ __forceinline__ v8f wmma16(v16h a, v16h b, v8f c) {
  return __builtin_amdgcn_wmma_f32_16x16x32_f16(false, a, false, b, (short)0, c, false, false);
}
__device__ __forceinline__ v8f zero8() { v8f z = {0.f,0.f,0.f,0.f,0.f,0.f,0.f,0.f}; return z; }

// CDNA5 async copy: 16 bytes global -> LDS per lane, tracked by ASYNCcnt.
__device__ __forceinline__ void async_cp16(unsigned ldsAddr, const void* gptr) {
  asm volatile("global_load_async_to_lds_b128 %0, %1, off"
               :: "v"(ldsAddr), "v"(gptr)
               : "memory");
}
__device__ __forceinline__ void wait_async0() {
  asm volatile("s_wait_asynccnt 0" ::: "memory");
}
__device__ __forceinline__ unsigned lds_off(const void* p) {
  return (unsigned)(unsigned long long)p;   // low 32 bits of flat LDS aperture addr
}

// ---------------------------------------------------------------------------
// Tiled WMMA GEMM: C[M,N] = A16[M,K] * Bt16[N,K]^T (+bias)(relu)(+residual)
// 256 threads = 8 waves; WG tile 128x256; wave tile 64x64 (16 WMMA / K-step);
// double-buffered async global->LDS staging, one barrier per K-step of 32.
// ---------------------------------------------------------------------------
#define LDT 40   // LDS row stride in halves (32 + 8 pad, 80B rows, 16B aligned)

__global__ __launch_bounds__(256) void k_gemm(
    const _Float16* __restrict__ A, const _Float16* __restrict__ Bt,
    int M, int N, int K,
    const float* __restrict__ bias, int doRelu,
    const float* __restrict__ residual,
    float* __restrict__ outF, _Float16* __restrict__ outH)
{
  __shared__ __align__(16) _Float16 sA[2][128 * LDT];
  __shared__ __align__(16) _Float16 sB[2][256 * LDT];
  const int tid  = threadIdx.x;
  const int wave = tid >> 5;
  const int lane = tid & 31;
  const int bm = blockIdx.x * 128;
  const int bn = blockIdx.y * 256;
  const int wm = (wave & 1) * 64;
  const int wn = (wave >> 1) * 64;

  v8f c[4][4];
#pragma unroll
  for (int i = 0; i < 4; ++i)
#pragma unroll
    for (int j = 0; j < 4; ++j) c[i][j] = zero8();

  const int srow   = tid >> 2;        // 0..63
  const int schunk = (tid & 3) * 8;   // halves: 0,8,16,24 (16B chunks)

  auto stage = [&](int buf, int kk) {
    const _Float16* ga = A + (size_t)(bm + srow) * K + kk + schunk;
    unsigned la = lds_off(&sA[buf][srow * LDT + schunk]);
    async_cp16(la, ga);
    async_cp16(la + 64u * LDT * 2, ga + (size_t)64 * K);
    const _Float16* gb = Bt + (size_t)(bn + srow) * K + kk + schunk;
    unsigned lb = lds_off(&sB[buf][srow * LDT + schunk]);
#pragma unroll
    for (int r = 0; r < 4; ++r)
      async_cp16(lb + (unsigned)(r * 64) * LDT * 2, gb + (size_t)(r * 64) * K);
  };

  stage(0, 0);
  int cur = 0;
  for (int k0 = 0; k0 < K; k0 += 32) {
    wait_async0();
    __syncthreads();                       // staged tile visible, prev reads done
    if (k0 + 32 < K) stage(cur ^ 1, k0 + 32);

    const _Float16* pa = sA[cur];
    const _Float16* pb = sB[cur];
    v16h a0 = frag_ld_a(pa + (wm + 0)  * LDT, LDT);
    v16h a1 = frag_ld_a(pa + (wm + 16) * LDT, LDT);
    v16h a2 = frag_ld_a(pa + (wm + 32) * LDT, LDT);
    v16h a3 = frag_ld_a(pa + (wm + 48) * LDT, LDT);
    v16h b0 = frag_ld_b(pb + (wn + 0)  * LDT, LDT);
    v16h b1 = frag_ld_b(pb + (wn + 16) * LDT, LDT);
    v16h b2 = frag_ld_b(pb + (wn + 32) * LDT, LDT);
    v16h b3 = frag_ld_b(pb + (wn + 48) * LDT, LDT);
    c[0][0] = wmma16(a0, b0, c[0][0]);
    c[0][1] = wmma16(a0, b1, c[0][1]);
    c[0][2] = wmma16(a0, b2, c[0][2]);
    c[0][3] = wmma16(a0, b3, c[0][3]);
    c[1][0] = wmma16(a1, b0, c[1][0]);
    c[1][1] = wmma16(a1, b1, c[1][1]);
    c[1][2] = wmma16(a1, b2, c[1][2]);
    c[1][3] = wmma16(a1, b3, c[1][3]);
    c[2][0] = wmma16(a2, b0, c[2][0]);
    c[2][1] = wmma16(a2, b1, c[2][1]);
    c[2][2] = wmma16(a2, b2, c[2][2]);
    c[2][3] = wmma16(a2, b3, c[2][3]);
    c[3][0] = wmma16(a3, b0, c[3][0]);
    c[3][1] = wmma16(a3, b1, c[3][1]);
    c[3][2] = wmma16(a3, b2, c[3][2]);
    c[3][3] = wmma16(a3, b3, c[3][3]);
    cur ^= 1;
  }

  // Epilogue: C layout -> row = base + e + 8*(lane>>4), col = base + (lane&15)
  const int c15 = lane & 15;
  const int hi  = lane >> 4;
#pragma unroll
  for (int i = 0; i < 4; ++i) {
    const int rb = bm + wm + i * 16 + 8 * hi;
#pragma unroll
    for (int j = 0; j < 4; ++j) {
      const int col = bn + wn + j * 16 + c15;
      const float bv = bias ? bias[col] : 0.f;
#pragma unroll
      for (int e = 0; e < 8; ++e) {
        float v = c[i][j][e] + bv;
        if (doRelu) v = fmaxf(v, 0.f);
        const size_t off = (size_t)(rb + e) * N + col;
        if (residual) v += residual[off];
        if (outF) outF[off] = v;
        if (outH) outH[off] = (_Float16)v;
      }
    }
  }
}

// ---------------------------------------------------------------------------
// Flash-style attention. One WG (4 waves) per (b, h, 64-query block).
// Each wave owns 16 queries; loops over 32 keys/iter with online softmax.
// Double-buffered async staging of K/V tiles; one barrier per iteration.
// qkv: [TOK][1536] f16 (Q cols 0..511 = h*64+dk, K cols 512.., V cols 1024..)
// vt:  [512][TOK] f16  (V transposed: row = h*64+dk)
// heads: [TOK][512] f16 output (col = h*64+dk)
// ---------------------------------------------------------------------------
#define LKS 72   // sK row stride (64 + 8)
#define LVS 40   // sV / sP row stride (32 + 8)

__global__ __launch_bounds__(128) void k_attn(
    const _Float16* __restrict__ qkv,
    const _Float16* __restrict__ vt,
    _Float16* __restrict__ heads)
{
  __shared__ __align__(16) _Float16 sK[2][32 * LKS];
  __shared__ __align__(16) _Float16 sV[2][64 * LVS];
  __shared__ __align__(16) _Float16 sP[4][16 * LVS];

  const int tid  = threadIdx.x;
  const int wave = tid >> 5;
  const int lane = tid & 31;
  const int qblk = blockIdx.x & 7;
  const int hh   = (blockIdx.x >> 3) & 7;
  const int b    = blockIdx.x >> 6;
  const int q0   = qblk * 64 + wave * 16;
  const int tok0 = b * NNQ;

  // Q fragments (16 queries x 64 dk) held in registers for the whole loop
  const _Float16* qrow = qkv + (size_t)(tok0 + q0) * 1536 + hh * 64;
  v16h qa0 = frag_ld_a(qrow, 1536);        // dk 0..31
  v16h qa1 = frag_ld_a(qrow + 32, 1536);   // dk 32..63

  float m[8], l[8];
  v8f o[4];
#pragma unroll
  for (int e = 0; e < 8; ++e) { m[e] = -1e30f; l[e] = 0.f; }
#pragma unroll
  for (int f = 0; f < 4; ++f) o[f] = zero8();

  const int krow   = tid >> 3;        // 0..15
  const int kchunk = (tid & 7) * 8;   // 8 chunks x 16B = 128B row
  const int vrow   = tid >> 2;        // 0..31
  const int vchunk = (tid & 3) * 8;   // 4 chunks x 16B = 64B row
  const int c15 = lane & 15;
  const int hi  = lane >> 4;

  auto stageKV = [&](int buf, int kt) {
    const _Float16* gk = qkv + (size_t)(tok0 + kt + krow) * 1536 + 512 + hh * 64 + kchunk;
    unsigned lk = lds_off(&sK[buf][krow * LKS + kchunk]);
    async_cp16(lk, gk);
    async_cp16(lk + 16u * LKS * 2, gk + (size_t)16 * 1536);
    const _Float16* gv = vt + (size_t)(hh * 64 + vrow) * TOK + tok0 + kt + vchunk;
    unsigned lv = lds_off(&sV[buf][vrow * LVS + vchunk]);
    async_cp16(lv, gv);
    async_cp16(lv + 32u * LVS * 2, gv + (size_t)32 * TOK);
  };

  stageKV(0, 0);
  int cur = 0;
  for (int kt = 0; kt < NNQ; kt += 32) {
    wait_async0();
    __syncthreads();
    if (kt + 32 < NNQ) stageKV(cur ^ 1, kt + 32);
    const _Float16* Kt = sK[cur];
    const _Float16* Vt = sV[cur];

    // scores: S[16q x 32k] via 4 WMMAs (K-dim = dk = 64)
    v8f s0 = zero8(), s1 = zero8();
    s0 = wmma16(qa0, frag_ld_b(Kt, LKS), s0);
    s0 = wmma16(qa1, frag_ld_b(Kt + 32, LKS), s0);
    s1 = wmma16(qa0, frag_ld_b(Kt + 16 * LKS, LKS), s1);
    s1 = wmma16(qa1, frag_ld_b(Kt + 16 * LKS + 32, LKS), s1);

    // online softmax (rows live in e, cols striped over 16 lanes per half)
    float resc[8];
#pragma unroll
    for (int e = 0; e < 8; ++e) {
      float x0 = s0[e] * 0.125f, x1 = s1[e] * 0.125f;
      float mx = fmaxf(x0, x1);
      mx = fmaxf(mx, __shfl_xor(mx, 1, 32));
      mx = fmaxf(mx, __shfl_xor(mx, 2, 32));
      mx = fmaxf(mx, __shfl_xor(mx, 4, 32));
      mx = fmaxf(mx, __shfl_xor(mx, 8, 32));
      float mn = fmaxf(m[e], mx);
      resc[e] = __expf(m[e] - mn);
      m[e] = mn;
      float p0 = __expf(x0 - mn), p1 = __expf(x1 - mn);
      float rs = p0 + p1;
      rs += __shfl_xor(rs, 1, 32);
      rs += __shfl_xor(rs, 2, 32);
      rs += __shfl_xor(rs, 4, 32);
      rs += __shfl_xor(rs, 8, 32);
      l[e] = l[e] * resc[e] + rs;
      s0[e] = p0; s1[e] = p1;
    }
#pragma unroll
    for (int f = 0; f < 4; ++f)
#pragma unroll
      for (int e = 0; e < 8; ++e) o[f][e] *= resc[e];

    // bounce P through wave-private LDS to get A-fragment layout
    _Float16* pbw = sP[wave];
#pragma unroll
    for (int e = 0; e < 8; ++e) {
      const int rr = e + 8 * hi;
      pbw[rr * LVS + c15]      = (_Float16)s0[e];
      pbw[rr * LVS + 16 + c15] = (_Float16)s1[e];
    }
    asm volatile("s_wait_dscnt 0" ::: "memory");
    v16h pa = frag_ld_a(pbw, LVS);

    // O[16q x 64dk] += P[16x32] * V[32x64]  (4 WMMAs)
    o[0] = wmma16(pa, frag_ld_b(Vt + 0  * LVS, LVS), o[0]);
    o[1] = wmma16(pa, frag_ld_b(Vt + 16 * LVS, LVS), o[1]);
    o[2] = wmma16(pa, frag_ld_b(Vt + 32 * LVS, LVS), o[2]);
    o[3] = wmma16(pa, frag_ld_b(Vt + 48 * LVS, LVS), o[3]);
    cur ^= 1;
  }

  float inv[8];
#pragma unroll
  for (int e = 0; e < 8; ++e) inv[e] = 1.f / l[e];
#pragma unroll
  for (int f = 0; f < 4; ++f) {
    const int col = hh * 64 + f * 16 + c15;
#pragma unroll
    for (int e = 0; e < 8; ++e) {
      const int row = tok0 + q0 + e + 8 * hi;
      heads[(size_t)row * DD + col] = (_Float16)(o[f][e] * inv[e]);
    }
  }
}

// ---------------------------------------------------------------------------
// Small helper kernels
// ---------------------------------------------------------------------------
__global__ __launch_bounds__(256) void k_embed(
    const float* __restrict__ x, const float* __restrict__ We,
    const float* __restrict__ be, float* __restrict__ h, _Float16* __restrict__ h16)
{
  const size_t i = (size_t)blockIdx.x * 256 + threadIdx.x;   // TOK*DD
  const int d = (int)(i & 511);
  const size_t t = i >> 9;
  const float v = x[t * 2] * We[d * 2] + x[t * 2 + 1] * We[d * 2 + 1] + be[d];
  h[i] = v; h16[i] = (_Float16)v;
}

__global__ __launch_bounds__(256) void k_bn_stats(
    const float* __restrict__ x, float* __restrict__ stats)
{
  const int c = blockIdx.x;        // channel 0..511
  float s = 0.f, s2 = 0.f;
  for (int r = threadIdx.x; r < TOK; r += 256) {
    const float v = x[(size_t)r * DD + c];
    s += v; s2 += v * v;
  }
  __shared__ float rs[256], rq[256];
  rs[threadIdx.x] = s; rq[threadIdx.x] = s2;
  __syncthreads();
  for (int st = 128; st > 0; st >>= 1) {
    if (threadIdx.x < st) { rs[threadIdx.x] += rs[threadIdx.x + st]; rq[threadIdx.x] += rq[threadIdx.x + st]; }
    __syncthreads();
  }
  if (threadIdx.x == 0) {
    const float mu = rs[0] * (1.f / TOK);
    const float var = rq[0] * (1.f / TOK) - mu * mu;
    stats[c] = mu;
    stats[DD + c] = rsqrtf(var + 1e-5f);
  }
}

__global__ __launch_bounds__(256) void k_bn_apply(
    const float* __restrict__ x, const float* __restrict__ stats,
    const float* __restrict__ gamma, const float* __restrict__ beta,
    float* __restrict__ h, _Float16* __restrict__ h16)
{
  const size_t i = (size_t)blockIdx.x * 256 + threadIdx.x;
  const int c = (int)(i & 511);
  const float v = (x[i] - stats[c]) * stats[DD + c] * gamma[c] + beta[c];
  h[i] = v; h16[i] = (_Float16)v;
}

__global__ __launch_bounds__(256) void k_transpose_v(
    const _Float16* __restrict__ qkv, _Float16* __restrict__ vtb)
{
  const size_t i = (size_t)blockIdx.x * 256 + threadIdx.x;  // TOK*DD
  const int c = (int)(i & 511);
  const size_t t = i >> 9;
  vtb[(size_t)c * TOK + t] = qkv[t * 1536 + 1024 + c];
}

__global__ __launch_bounds__(256) void k_prep_qkv(
    const float* __restrict__ Wq, const float* __restrict__ Wk,
    const float* __restrict__ Wv, _Float16* __restrict__ qkvt)
{
  const size_t i = (size_t)blockIdx.x * 256 + threadIdx.x;  // L*1536*512
  const int k = (int)(i & 511);
  const size_t r = i >> 9;
  const int n  = (int)(r % 1536);
  const int li = (int)(r / 1536);
  const float* W; int nn;
  if (n < 512)       { W = Wq; nn = n; }
  else if (n < 1024) { W = Wk; nn = n - 512; }
  else               { W = Wv; nn = n - 1024; }
  const int hh = nn >> 6, dk = nn & 63;
  qkvt[i] = (_Float16)W[(((size_t)li * 8 + hh) * 512 + k) * 64 + dk];
}

__global__ __launch_bounds__(256) void k_prep_wo(
    const float* __restrict__ Wo, _Float16* __restrict__ wot)
{
  const size_t i = (size_t)blockIdx.x * 256 + threadIdx.x;  // L*512*512
  const int k = (int)(i & 511);
  const size_t r = i >> 9;
  const int n  = (int)(r % 512);
  const int li = (int)(r / 512);
  const int hh = k >> 6, dk = k & 63;
  wot[i] = (_Float16)Wo[(((size_t)li * 8 + hh) * 64 + dk) * 512 + n];
}

__global__ __launch_bounds__(256) void k_cvt(
    const float* __restrict__ in, _Float16* __restrict__ out)
{
  const size_t i = (size_t)blockIdx.x * 256 + threadIdx.x;
  out[i] = (_Float16)in[i];
}

__global__ __launch_bounds__(256) void k_mean(
    const float* __restrict__ h, float* __restrict__ out)
{
  const int i = blockIdx.x * 256 + threadIdx.x;   // B*DD
  const int d = i & 511;
  const int b = i >> 9;
  float s = 0.f;
  for (int n = 0; n < NNQ; ++n) s += h[((size_t)b * NNQ + n) * DD + d];
  out[i] = s * (1.f / NNQ);
}

// ---------------------------------------------------------------------------
extern "C" void kernel_launch(void* const* d_in, const int* in_sizes, int n_in,
                              void* d_out, int out_size, void* d_ws, size_t ws_size,
                              hipStream_t stream) {
  (void)in_sizes; (void)n_in; (void)out_size; (void)ws_size;
  const float* x   = (const float*)d_in[0];
  const float* We  = (const float*)d_in[1];
  const float* be  = (const float*)d_in[2];
  const float* Wq  = (const float*)d_in[3];
  const float* Wk  = (const float*)d_in[4];
  const float* Wv  = (const float*)d_in[5];
  const float* Wo  = (const float*)d_in[6];
  const float* g1  = (const float*)d_in[7];
  const float* b1  = (const float*)d_in[8];
  const float* W1  = (const float*)d_in[9];
  const float* bb1 = (const float*)d_in[10];
  const float* W2  = (const float*)d_in[11];
  const float* bb2 = (const float*)d_in[12];
  const float* g2  = (const float*)d_in[13];
  const float* b2  = (const float*)d_in[14];
  float* out = (float*)d_out;

  char* ws = (char*)d_ws;
  size_t off = 0;
  auto take = [&](size_t bytes) -> char* {
    char* p = ws + off;
    off = (off + bytes + 255) & ~(size_t)255;
    return p;
  };
  float*    hbuf    = (float*)   take((size_t)TOK * DD * 4);
  _Float16* h16     = (_Float16*)take((size_t)TOK * DD * 2);
  _Float16* qkv16   = (_Float16*)take((size_t)TOK * 1536 * 2);
  _Float16* vt16    = (_Float16*)take((size_t)DD * TOK * 2);
  _Float16* heads16 = (_Float16*)take((size_t)TOK * DD * 2);
  float*    prebn   = (float*)   take((size_t)TOK * DD * 4);
  float*    stats   = (float*)   take(1024 * 4);
  _Float16* qkvt    = (_Float16*)take((size_t)LLN * 1536 * DD * 2);
  _Float16* wot     = (_Float16*)take((size_t)LLN * DD * DD * 2);
  _Float16* w1h     = (_Float16*)take((size_t)LLN * FFD * DD * 2);
  _Float16* w2h     = (_Float16*)take((size_t)LLN * DD * FFD * 2);
  _Float16* ff16    = qkv16;  // FFN activations reuse the (dead) qkv+vt region (64 MiB)

  // ---- weight prep (fp32 -> fp16, [N][K] layouts) ----
  k_prep_qkv<<<(LLN * 1536 * 512) / 256, 256, 0, stream>>>(Wq, Wk, Wv, qkvt);
  k_prep_wo <<<(LLN * 512 * 512) / 256, 256, 0, stream>>>(Wo, wot);
  k_cvt     <<<(LLN * FFD * DD) / 256, 256, 0, stream>>>(W1, w1h);   // W1: [FF][D] already
  k_cvt     <<<(LLN * DD * FFD) / 256, 256, 0, stream>>>(W2, w2h);   // W2: [D][FF] already

  // ---- embed ----
  k_embed<<<(TOK * DD) / 256, 256, 0, stream>>>(x, We, be, hbuf, h16);

  for (int i = 0; i < LLN; ++i) {
    const _Float16* qkvt_i = qkvt + (size_t)i * 1536 * DD;
    const _Float16* wot_i  = wot  + (size_t)i * DD * DD;
    const _Float16* w1_i   = w1h  + (size_t)i * FFD * DD;
    const _Float16* w2_i   = w2h  + (size_t)i * DD * FFD;

    // QKV projection: [TOK,512] x [512,1536]
    k_gemm<<<dim3(TOK / 128, 1536 / 256), 256, 0, stream>>>(
        h16, qkvt_i, TOK, 1536, DD, nullptr, 0, nullptr, nullptr, qkv16);
    k_transpose_v<<<(TOK * DD) / 256, 256, 0, stream>>>(qkv16, vt16);
    // attention
    k_attn<<<BB * HHN * (NNQ / 64), 128, 0, stream>>>(qkv16, vt16, heads16);
    // output projection + residual
    k_gemm<<<dim3(TOK / 128, DD / 256), 256, 0, stream>>>(
        heads16, wot_i, TOK, DD, DD, nullptr, 0, hbuf, prebn, nullptr);
    // BN1
    k_bn_stats<<<DD, 256, 0, stream>>>(prebn, stats);
    k_bn_apply<<<(TOK * DD) / 256, 256, 0, stream>>>(
        prebn, stats, g1 + (size_t)i * DD, b1 + (size_t)i * DD, hbuf, h16);
    // FFN
    k_gemm<<<dim3(TOK / 128, FFD / 256), 256, 0, stream>>>(
        h16, w1_i, TOK, FFD, DD, bb1 + (size_t)i * FFD, 1, nullptr, nullptr, ff16);
    k_gemm<<<dim3(TOK / 128, DD / 256), 256, 0, stream>>>(
        ff16, w2_i, TOK, DD, FFD, bb2 + (size_t)i * DD, 0, hbuf, prebn, nullptr);
    // BN2
    k_bn_stats<<<DD, 256, 0, stream>>>(prebn, stats);
    k_bn_apply<<<(TOK * DD) / 256, 256, 0, stream>>>(
        prebn, stats, g2 + (size_t)i * DD, b2 + (size_t)i * DD, hbuf, h16);
  }

  // outputs: h [B,N,D] then mean over N [B,D]
  hipMemcpyAsync(d_out, hbuf, (size_t)TOK * DD * 4, hipMemcpyDeviceToDevice, stream);
  k_mean<<<(BB * DD) / 256, 256, 0, stream>>>(hbuf, out + (size_t)TOK * DD);
}
